// MSTGCN_block_57200374448664
// MI455X (gfx1250) — matrix operand order repaired and checked by reading, not compiled, not run on hardware
//
#include <hip/hip_runtime.h>

typedef __attribute__((ext_vector_type(16))) __bf16 v16bf;
typedef __attribute__((ext_vector_type(8)))  float  v8f;

namespace {
constexpr int Bc = 32, Nc = 1024, Fc = 3, Tc = 32, Kc = 3, Cc = 64;
constexpr int FT  = 96;          // Fc*Tc
constexpr int MC  = 64;          // m rows staged per chunk
constexpr int NCH = Nc / MC;     // 16 chunks
constexpr int NT  = 16;          // nodes per block

// LDS byte offsets (S buffer overlays the phase-A staging buffers)
constexpr int OFF_CS = 0;                 // bf16 [3][2][32][16]  chebT, WMMA-A layout (6144 B)
constexpr int OFF_XS = 6144;              // bf16 [2][6][32][16]  x, WMMA-B layout     (12288 B)
constexpr int OFF_SW = 0;                 // bf16 [8 waves][64][36] S w/ t-halo        (36864 B)
constexpr int OFF_Z  = 36864;             // f32  [3][16][96]                          (18432 B)
constexpr int OFF_WA = OFF_Z  + 18432;    // bf16 [3 tap][2 kblk][4 mt][32][16]        (24576 B)
constexpr int OFF_TH = OFF_WA + 24576;    // f32  [9][64]                              (2304 B)
constexpr int OFF_BT = OFF_TH + 2304;     // f32  [64]
constexpr int OFF_BR = OFF_BT + 256;      // f32  [64]
constexpr int OFF_WR = OFF_BR + 256;      // f32  [64][3]
constexpr int OFF_GM = OFF_WR + 768;      // f32  [64]
constexpr int OFF_BE = OFF_GM + 256;      // f32  [64]
constexpr int LDS_BYTES = OFF_BE + 256;   // 83968 B (< 320 KB/WG on gfx1250)
}

__device__ __forceinline__ __bf16 f2bf(float f) {
  unsigned u = __builtin_bit_cast(unsigned, f);
  unsigned r = u + 0x7fffu + ((u >> 16) & 1u);   // round-to-nearest-even
  return __builtin_bit_cast(__bf16, (unsigned short)(r >> 16));
}

extern "C" __global__ __launch_bounds__(256, 1)
void mstgcn_fused(const float* __restrict__ x,    const float* __restrict__ cheb,
                  const float* __restrict__ theta,const float* __restrict__ wt,
                  const float* __restrict__ btv,  const float* __restrict__ wrv,
                  const float* __restrict__ brv,  const float* __restrict__ gmv,
                  const float* __restrict__ bev,  float* __restrict__ out)
{
  __shared__ __align__(16) unsigned char lds[LDS_BYTES];
  __bf16* c_s  = (__bf16*)(lds + OFF_CS);
  __bf16* x_s  = (__bf16*)(lds + OFF_XS);
  __bf16* s_s  = (__bf16*)(lds + OFF_SW);
  float*  z_s  = (float*) (lds + OFF_Z);
  __bf16* w_a  = (__bf16*)(lds + OFF_WA);
  float*  th_s = (float*) (lds + OFF_TH);
  float*  bt_s = (float*) (lds + OFF_BT);
  float*  br_s = (float*) (lds + OFF_BR);
  float*  wr_s = (float*) (lds + OFF_WR);
  float*  gm_s = (float*) (lds + OFF_GM);
  float*  be_s = (float*) (lds + OFF_BE);

  const int tid  = threadIdx.x;
  const int lane = tid & 31;
  const int wv   = __builtin_amdgcn_readfirstlane(tid >> 5);  // scalar wave id
  const int bb = blockIdx.y;
  const int n0 = blockIdx.x * NT;

  // ---- stage static operands: w_time into WMMA-A layout, theta, vectors ----
  for (int idx = tid; idx < Cc * Cc * 3; idx += 256) {   // 12288 elems
    int co  = idx / (Cc * 3);
    int r   = idx % (Cc * 3);
    int ci  = r / 3;
    int tap = r % 3;
    float v = wt[idx];
    int mt  = co >> 4, row = co & 15;
    int kb  = ci >> 5, Kk  = ci & 31;
    int ln  = row + (((Kk >> 4) & 1) << 4);
    int j   = Kk & 15;
    w_a[(((tap * 2 + kb) * 4 + mt) * 32 + ln) * 16 + j] = f2bf(v);
  }
  for (int idx = tid; idx < 9 * 64; idx += 256) th_s[idx] = theta[idx];
  if (tid < 64) { bt_s[tid] = btv[tid]; br_s[tid] = brv[tid];
                  gm_s[tid] = gmv[tid]; be_s[tid] = bev[tid]; }
  if (tid < 192) wr_s[tid] = wrv[tid];

  // ---- per-thread loop-invariant staging coords (all shifts/ands) ----------
  // x: thread owns row mmx = tid>>2, 24 consecutive cols from (tid&3)*24
  const int mmx = tid >> 2;
  const int ftb = (tid & 3) * 24;
  const int Kx  = mmx & 31, mbx = mmx >> 5;
  const int lnHi = ((Kx >> 3) & 1) << 4;
  const int jx   = (Kx & 7) | (((Kx >> 4) & 1) << 3);
  // cheb: waves 0..5 (threads 0..191) each own one (k,mm) row: 16 n = 4x float4
  const int kc  = tid >> 6;            // 0..2 (tid<192)
  const int mmc = tid & 63;
  const int Kc2 = mmc & 31, mbc = mmc >> 5;
  const int jc  = (Kc2 & 7) | (((Kc2 >> 4) & 1) << 3);
  const int lnC = ((Kc2 >> 3) & 1) << 4;
  const float4* crow0 = (const float4*)(cheb + (size_t)kc * Nc * Nc
                                             + (size_t)mmc * Nc + n0);
  __bf16* cdst = c_s + ((kc * 2 + mbc) * 32 + lnC) * 16 + jc;

  // ================= Phase A: Z_k = T_k^T @ X_b  (bf16 WMMA, f32 acc) ======
  // Branch-free tile assignment: wave wv owns tiles {wv, wv+8, min(wv+16,17)}.
  // Waves 2..7 redundantly recompute tile 17 (identical data -> benign dup).
  const int tile0 = wv;
  const int tile1 = wv + 8;
  const int tile2 = (wv + 16 < 18) ? (wv + 16) : 17;
  const int k0 = tile0 / 6, f0 = tile0 % 6;
  const int k1 = tile1 / 6, f1 = tile1 % 6;
  const int k2 = tile2 / 6, f2 = tile2 % 6;

  v8f zero8 = {};
  v8f acc0 = zero8, acc1 = zero8, acc2 = zero8;

  for (int mc = 0; mc < NCH; ++mc) {
    const int m0 = mc * MC;
    // chebT chunk -> A layout: A[n][m]; 4x global_load_b128 per owning thread.
    // Guard is SCALAR (wv<6 == tid<192): pure s_cbranch, no EXEC churn.
    if (wv < 6) {
      const float4* crow = crow0 + (size_t)(m0 / 4) * (Nc / 4) * 4; // + m0*Nc floats
      #pragma unroll
      for (int q = 0; q < 4; ++q) {
        float4 v4 = crow[q];
        cdst[(q * 4 + 0) * 16] = f2bf(v4.x);
        cdst[(q * 4 + 1) * 16] = f2bf(v4.y);
        cdst[(q * 4 + 2) * 16] = f2bf(v4.z);
        cdst[(q * 4 + 3) * 16] = f2bf(v4.w);
      }
    }
    // x chunk -> B layout: B[m][ft], col = ft%16, K = m%32
    const float* xrow = x + (size_t)(bb * Nc + m0 + mmx) * FT + ftb;
    #pragma unroll
    for (int j2 = 0; j2 < 24; ++j2) {
      float v = xrow[j2];
      int ft = ftb + j2;
      int ftile = ft >> 4, col = ft & 15;
      x_s[((mbx * 6 + ftile) * 32 + (col + lnHi)) * 16 + jx] = f2bf(v);
    }
    if (mc + 1 < NCH)
      __builtin_prefetch(xrow + (size_t)MC * FT, 0, 0);
    __syncthreads();

    #pragma unroll
    for (int mb = 0; mb < 2; ++mb) {
      v16bf a0 = *(const v16bf*)(c_s + ((k0 * 2 + mb) * 32 + lane) * 16);
      v16bf b0 = *(const v16bf*)(x_s + ((mb * 6 + f0) * 32 + lane) * 16);
      acc0 = __builtin_amdgcn_wmma_f32_16x16x32_bf16(false, a0, false, b0,
                                                     (short)0, acc0, false, false);
      v16bf a1 = *(const v16bf*)(c_s + ((k1 * 2 + mb) * 32 + lane) * 16);
      v16bf b1 = *(const v16bf*)(x_s + ((mb * 6 + f1) * 32 + lane) * 16);
      acc1 = __builtin_amdgcn_wmma_f32_16x16x32_bf16(false, a1, false, b1,
                                                     (short)0, acc1, false, false);
      v16bf a2 = *(const v16bf*)(c_s + ((k2 * 2 + mb) * 32 + lane) * 16);
      v16bf b2 = *(const v16bf*)(x_s + ((mb * 6 + f2) * 32 + lane) * 16);
      acc2 = __builtin_amdgcn_wmma_f32_16x16x32_bf16(false, a2, false, b2,
                                                     (short)0, acc2, false, false);
    }
    __syncthreads();
  }

  // spill Z (pre-theta) to LDS; C/D layout: M = v + 8*(lane/16), N = lane%16
  {
    const int rowg = lane >> 4, col = lane & 15;
    #pragma unroll
    for (int v = 0; v < 8; ++v) {
      z_s[(k0 * NT + (v + 8 * rowg)) * FT + f0 * 16 + col] = acc0[v];
      z_s[(k1 * NT + (v + 8 * rowg)) * FT + f1 * 16 + col] = acc1[v];
      z_s[(k2 * NT + (v + 8 * rowg)) * FT + f2 * 16 + col] = acc2[v];
    }
  }
  __syncthreads();

  // ============ Phase B+C per wave: theta+ReLU, time conv WMMA, LN =========
  __bf16* sw = s_s + wv * (64 * 36);
  const float inv64 = 1.0f / 64.0f;

  #pragma unroll 1
  for (int ni = 0; ni < 2; ++ni) {
    const int nn  = wv * 2 + ni;
    const int ngl = n0 + nn;

    // zero the t-halo (conv padding), then S[ci][t=lane]
    for (int ci = lane; ci < 64; ci += 32) {
      sw[ci * 36 + 0]  = f2bf(0.f);
      sw[ci * 36 + 33] = f2bf(0.f);
    }
    float zf[9];
    #pragma unroll
    for (int i = 0; i < 9; ++i)
      zf[i] = z_s[((i / 3) * NT + nn) * FT + (i % 3) * Tc + lane];
    for (int ci = 0; ci < 64; ++ci) {
      float s = 0.f;
      #pragma unroll
      for (int i = 0; i < 9; ++i) s = fmaf(zf[i], th_s[i * 64 + ci], s);
      sw[ci * 36 + 1 + lane] = f2bf(fmaxf(s, 0.f));
    }
    asm volatile("s_wait_dscnt 0" ::: "memory");  // same-wave LDS W->R before WMMA

    const float* xr = x + (size_t)(bb * Nc + ngl) * FT;
    #pragma unroll 1
    for (int tt = 0; tt < 2; ++tt) {
      v8f oacc[4] = { zero8, zero8, zero8, zero8 };
      const int tcol = tt * 16 + (lane & 15);
      const int rowg = lane >> 4;
      #pragma unroll
      for (int kb = 0; kb < 2; ++kb) {
        #pragma unroll
        for (int tap = 0; tap < 3; ++tap) {
          v16bf bm;
          #pragma unroll
          for (int j = 0; j < 16; ++j)
            bm[j] = sw[(kb * 32 + rowg * 16 + j) * 36 + tcol + tap];
          #pragma unroll
          for (int mt = 0; mt < 4; ++mt) {
            v16bf am = *(const v16bf*)(w_a + (((tap * 2 + kb) * 4 + mt) * 32 + lane) * 16);
            oacc[mt] = __builtin_amdgcn_wmma_f32_16x16x32_bf16(
                false, am, false, bm, (short)0, oacc[mt], false, false);
          }
        }
      }
      // epilogue: bias + 1x1 residual + ReLU + LayerNorm over 64 channels
      float x0 = xr[0 * Tc + tcol], x1 = xr[1 * Tc + tcol], x2 = xr[2 * Tc + tcol];
      float vals[32];
      float sum = 0.f, ss = 0.f;
      #pragma unroll
      for (int mt = 0; mt < 4; ++mt) {
        #pragma unroll
        for (int v = 0; v < 8; ++v) {
          int c = mt * 16 + rowg * 8 + v;
          float r = oacc[mt][v] + bt_s[c] + br_s[c]
                  + wr_s[c * 3 + 0] * x0 + wr_s[c * 3 + 1] * x1 + wr_s[c * 3 + 2] * x2;
          r = fmaxf(r, 0.f);
          vals[mt * 8 + v] = r;
          sum += r; ss += r * r;
        }
      }
      sum += __shfl_xor(sum, 16, 32);   // partner lane holds the other 32 channels
      ss  += __shfl_xor(ss, 16, 32);
      float mu  = sum * inv64;
      float var = ss * inv64 - mu * mu;
      float rin = rsqrtf(var + 1e-5f);
      float* og = out + (size_t)(bb * Nc + ngl) * Cc * Tc;
      #pragma unroll
      for (int mt = 0; mt < 4; ++mt) {
        #pragma unroll
        for (int v = 0; v < 8; ++v) {
          int c = mt * 16 + rowg * 8 + v;
          og[(size_t)c * Tc + tcol] = (vals[mt * 8 + v] - mu) * rin * gm_s[c] + be_s[c];
        }
      }
    }
  }
}

extern "C" void kernel_launch(void* const* d_in, const int* in_sizes, int n_in,
                              void* d_out, int out_size, void* d_ws, size_t ws_size,
                              hipStream_t stream) {
  const float* x     = (const float*)d_in[0];
  const float* cheb  = (const float*)d_in[1];
  const float* theta = (const float*)d_in[2];
  const float* wt    = (const float*)d_in[3];
  const float* bt    = (const float*)d_in[4];
  const float* wr    = (const float*)d_in[5];
  const float* br    = (const float*)d_in[6];
  const float* gm    = (const float*)d_in[7];
  const float* be    = (const float*)d_in[8];
  float* out = (float*)d_out;
  dim3 grid(Nc / NT, Bc);   // 64 node-tiles x 32 batches = 2048 blocks
  mstgcn_fused<<<grid, dim3(256), 0, stream>>>(x, cheb, theta, wt, bt, wr, br, gm, be, out);
  (void)in_sizes; (void)n_in; (void)out_size; (void)d_ws; (void)ws_size;
}
